// SelfAttention_51316269253024
// MI455X (gfx1250) — matrix operand order repaired and checked by reading, not compiled, not run on hardware
//
#include <hip/hip_runtime.h>
#include <stdint.h>

// ---------------------------------------------------------------- types
typedef __attribute__((ext_vector_type(16))) __bf16 v16bf;
typedef __attribute__((ext_vector_type(8)))  float  v8f;
typedef __attribute__((ext_vector_type(4)))  unsigned int u32x4;
typedef __attribute__((ext_vector_type(8)))  int    i32x8;
typedef __attribute__((ext_vector_type(4)))  int    i32x4;

#define T_SEQ   2048
#define D_MODEL 2048
#define NHEAD   16
#define HDIM    128
#define D3      6144

#if __has_builtin(__builtin_amdgcn_tensor_load_to_lds) && __has_builtin(__builtin_amdgcn_s_wait_tensorcnt)
#define HAVE_TDM 1
#else
#define HAVE_TDM 0
#endif

// ---------------------------------------------------------------- helpers
__device__ __forceinline__ unsigned short f2bf(float x) {
  __bf16 h = (__bf16)x;
  return __builtin_bit_cast(unsigned short, h);
}

__device__ __forceinline__ v8f wmma_bf16(v16bf a, v16bf b, v8f c) {
  return __builtin_amdgcn_wmma_f32_16x16x32_bf16(false, a, false, b, (short)0, c, false, false);
}

__device__ __forceinline__ unsigned lds_off(const void* p) {
  // generic pointer to LDS: low 32 bits are the LDS byte offset
  return (unsigned)(unsigned long long)p;
}

#if HAVE_TDM
// 2D bf16 tile load, tile_d0 elements contiguous (innermost), tile_d1 lines,
// line stride stride0 (elements). Emits tensor_load_to_lds (TENSORcnt).
__device__ __forceinline__ void tdm_load_2d_bf16(unsigned ldsoff, const void* gaddr,
                                                 unsigned tile_d0, unsigned tile_d1,
                                                 unsigned tensor_d0, unsigned tensor_d1,
                                                 unsigned long long stride0) {
  unsigned long long ga = (unsigned long long)gaddr;
  u32x4 g0;
  g0[0] = 1u;                                            // count=1, user mode
  g0[1] = ldsoff;                                        // lds_addr
  g0[2] = (unsigned)(ga & 0xFFFFFFFFu);                  // global_addr[31:0]
  g0[3] = (unsigned)((ga >> 32) & 0x1FFFFFFu) | (2u << 30); // addr[56:32] | type=2
  i32x8 g1;
  g1[0] = (int)(1u << 16);                               // data_size=1 (2B)
  g1[1] = (int)((tensor_d0 & 0xFFFFu) << 16);            // tensor_dim0[15:0]
  g1[2] = (int)(((tensor_d0 >> 16) & 0xFFFFu) | ((tensor_d1 & 0xFFFFu) << 16));
  g1[3] = (int)(((tensor_d1 >> 16) & 0xFFFFu) | (tile_d0 << 16));
  g1[4] = (int)(tile_d1 & 0xFFFFu);                      // tile_dim1, tile_dim2=0
  g1[5] = (int)(stride0 & 0xFFFFFFFFu);                  // dim0 stride low
  g1[6] = (int)((stride0 >> 32) & 0xFFFFu);              // dim0 stride high, stride1=0
  g1[7] = 0;
  i32x4 z4 = {0, 0, 0, 0};
#if defined(__clang_major__) && (__clang_major__ >= 23)
  i32x8 z8 = {0, 0, 0, 0, 0, 0, 0, 0};
  __builtin_amdgcn_tensor_load_to_lds(g0, g1, z4, z4, z8, 0);
#else
  __builtin_amdgcn_tensor_load_to_lds(g0, g1, z4, z4, 0);
#endif
}
#endif

// cooperative tile copy fallback (also used if TDM builtin is unavailable)
__device__ __forceinline__ void copy_tile_coop(unsigned short* dst, const unsigned short* src,
                                               int rows, int rowElems, int srcStride,
                                               int tid, int nthreads) {
  int cpr = rowElems >> 3;            // 16B chunks per row
  int total = rows * cpr;
  for (int c = tid; c < total; c += nthreads) {
    int r = c / cpr;
    int o = (c - r * cpr) << 3;
    *(uint4*)(dst + (size_t)r * rowElems + o) = *(const uint4*)(src + (size_t)r * srcStride + o);
  }
}

// A fragment: 16x32 bf16, row-major source with given stride.
// lanes 0-15: K 0-15 ; lanes 16-31: K 16-31
__device__ __forceinline__ v16bf load_A16(const unsigned short* base, int row0, int koff,
                                          int stride, int lane) {
  int r  = row0 + (lane & 15);
  int kh = ((lane >> 4) & 1) * 16;
  const unsigned short* p = base + (size_t)r * stride + koff + kh;
  union { v16bf v; unsigned u[8]; } f;
#pragma unroll
  for (int i = 0; i < 8; ++i) f.u[i] = *(const unsigned*)(p + 2 * i);
  return f.v;
}

// B fragment where source row n holds contiguous K: B[kk,n] = base[n*stride + kk]
// lanes 0-15: K 0-15 ; lanes 16-31: K 16-31
__device__ __forceinline__ v16bf load_rowB16(const unsigned short* base, int n, int koff,
                                             int stride, int lane) {
  int kh = ((lane >> 4) & 1) * 16;
  const unsigned short* p = base + (size_t)n * stride + koff + kh;
  union { v16bf v; unsigned u[8]; } f;
#pragma unroll
  for (int i = 0; i < 8; ++i) f.u[i] = *(const unsigned*)(p + 2 * i);
  return f.v;
}

// B fragment where source is row-major over K rows: B[kk,n] = base[kk*rowstride + n]
__device__ __forceinline__ v16bf load_colB16(const unsigned short* base, int n, int koff,
                                             int rowstride, int lane) {
  int kh = ((lane >> 4) & 1) * 16;
  union { v16bf v; unsigned short us[16]; } f;
#pragma unroll
  for (int e = 0; e < 16; ++e) f.us[e] = base[(size_t)(koff + kh + e) * rowstride + n];
  return f.v;
}

// ---------------------------------------------------------------- kernels
__global__ void cast_bf16_kernel(const float* __restrict__ src, unsigned* __restrict__ dst, int n2) {
  int i = blockIdx.x * blockDim.x + threadIdx.x;
  if (i < n2) {
    float a = src[2 * (size_t)i], b = src[2 * (size_t)i + 1];
    dst[i] = (unsigned)f2bf(a) | ((unsigned)f2bf(b) << 16);
  }
}

__global__ void rmsnorm_rows_kernel(const float* __restrict__ x, unsigned short* __restrict__ xn) {
  int row = blockIdx.x, tid = threadIdx.x;        // 256 threads, D=2048
  const float* xr = x + (size_t)row * D_MODEL;
  float v[8], ss = 0.f;
#pragma unroll
  for (int i = 0; i < 8; ++i) { v[i] = xr[tid + i * 256]; ss += v[i] * v[i]; }
#pragma unroll
  for (int m = 16; m >= 1; m >>= 1) ss += __shfl_xor(ss, m);
  __shared__ float wsum[8];
  int wid = tid >> 5, lane = tid & 31;
  if (lane == 0) wsum[wid] = ss;
  __syncthreads();
  float tot = 0.f;
#pragma unroll
  for (int i = 0; i < 8; ++i) tot += wsum[i];
  float sc = rsqrtf(tot * (1.0f / D_MODEL) + 1e-6f);
  unsigned short* o = xn + (size_t)row * D_MODEL;
#pragma unroll
  for (int i = 0; i < 8; ++i) o[tid + i * 256] = f2bf(v[i] * sc);
}

// C[MxN] = A[MxK] @ B[NxK]^T  (+ optional residual), 128x128 tile per WG, 8 waves
__global__ __launch_bounds__(256) void gemm_bf16_kernel(
    const unsigned short* __restrict__ A, const unsigned short* __restrict__ B,
    float* __restrict__ C, const float* __restrict__ resid, int M, int N, int K) {
  __shared__ unsigned short Abuf[2][128 * 32];
  __shared__ unsigned short Bbuf[2][128 * 32];
  int tid = threadIdx.x, wid = tid >> 5, lane = tid & 31;
  int m0 = blockIdx.x * 128, n0 = blockIdx.y * 128;
  int wm = wid >> 1, wn = wid & 1;                 // 4x2 wave grid, 32x64 per wave

  v8f zero = {0.f, 0.f, 0.f, 0.f, 0.f, 0.f, 0.f, 0.f};
  v8f acc[2][4];
#pragma unroll
  for (int i = 0; i < 2; ++i)
#pragma unroll
    for (int j = 0; j < 4; ++j) acc[i][j] = zero;

#if HAVE_TDM
  if (wid == 0) {
    tdm_load_2d_bf16(lds_off(&Abuf[0][0]), A + (size_t)m0 * K, 32, 128, (unsigned)K, (unsigned)M, (unsigned long long)K);
    tdm_load_2d_bf16(lds_off(&Bbuf[0][0]), B + (size_t)n0 * K, 32, 128, (unsigned)K, (unsigned)N, (unsigned long long)K);
  }
#endif
  for (int kk = 0, it = 0; kk < K; kk += 32, ++it) {
    int buf = it & 1;
#if HAVE_TDM
    if (wid == 0) {
      if (kk + 32 < K) {
        tdm_load_2d_bf16(lds_off(&Abuf[buf ^ 1][0]), A + (size_t)m0 * K + (kk + 32), 32, 128, (unsigned)K, (unsigned)M, (unsigned long long)K);
        tdm_load_2d_bf16(lds_off(&Bbuf[buf ^ 1][0]), B + (size_t)n0 * K + (kk + 32), 32, 128, (unsigned)K, (unsigned)N, (unsigned long long)K);
        __builtin_amdgcn_s_wait_tensorcnt(2);   // current buffer's 2 loads done
      } else {
        __builtin_amdgcn_s_wait_tensorcnt(0);
      }
    }
    __syncthreads();
#else
    copy_tile_coop(&Abuf[buf][0], A + (size_t)m0 * K + kk, 128, 32, K, tid, 256);
    copy_tile_coop(&Bbuf[buf][0], B + (size_t)n0 * K + kk, 128, 32, K, tid, 256);
    __syncthreads();
#endif
    v16bf a0 = load_A16(&Abuf[buf][0], wm * 32,      0, 32, lane);
    v16bf a1 = load_A16(&Abuf[buf][0], wm * 32 + 16, 0, 32, lane);
    v16bf b[4];
#pragma unroll
    for (int nt = 0; nt < 4; ++nt)
      b[nt] = load_rowB16(&Bbuf[buf][0], wn * 64 + nt * 16 + (lane & 15), 0, 32, lane);
#pragma unroll
    for (int nt = 0; nt < 4; ++nt) {
      acc[0][nt] = wmma_bf16(a0, b[nt], acc[0][nt]);
      acc[1][nt] = wmma_bf16(a1, b[nt], acc[1][nt]);
    }
    __syncthreads();
  }
  int half8 = ((lane >> 4) & 1) * 8, col_l = lane & 15;
#pragma unroll
  for (int mt = 0; mt < 2; ++mt)
#pragma unroll
    for (int nt = 0; nt < 4; ++nt)
#pragma unroll
      for (int r = 0; r < 8; ++r) {
        int row = m0 + wm * 32 + mt * 16 + r + half8;
        int col = n0 + wn * 64 + nt * 16 + col_l;
        size_t idx = (size_t)row * N + col;
        float val = acc[mt][nt][r];
        if (resid) val += resid[idx];
        C[idx] = val;
      }
}

// per-(t,head): rmsnorm(q),rmsnorm(k) over hd, RoPE, emit bf16 q/k/v + f32 k/v outputs
__global__ void rope_kernel(const float* __restrict__ qkv,
                            const float* __restrict__ cosb, const float* __restrict__ sinb,
                            unsigned short* __restrict__ qb, unsigned short* __restrict__ kb,
                            unsigned short* __restrict__ vb, float* __restrict__ outbase) {
  int head = blockIdx.x, t = blockIdx.y, tid = threadIdx.x;   // 128 threads
  const float* qp = qkv + (size_t)t * D3 + head * HDIM;
  const float* kp = qp + D_MODEL;
  const float* vp = qp + 2 * D_MODEL;
  float qv = qp[tid], kv = kp[tid];
  float sq = qv * qv, sk = kv * kv;
#pragma unroll
  for (int m = 16; m >= 1; m >>= 1) { sq += __shfl_xor(sq, m); sk += __shfl_xor(sk, m); }
  __shared__ float aq[4], ak[4];
  int wid = tid >> 5, lane = tid & 31;
  if (lane == 0) { aq[wid] = sq; ak[wid] = sk; }
  __syncthreads();
  float rq = rsqrtf((aq[0] + aq[1] + aq[2] + aq[3]) * (1.0f / HDIM) + 1e-6f);
  float rk = rsqrtf((ak[0] + ak[1] + ak[2] + ak[3]) * (1.0f / HDIM) + 1e-6f);
  const float attscale = 0.08838834764831845f;  // 1/sqrt(128), folded into q
  size_t obase = (size_t)t * D_MODEL + head * HDIM;
  if (tid < 64) {
    float x1 = qp[tid] * rq, x2 = qp[tid + 64] * rq;
    float c = cosb[(size_t)t * 64 + tid], s = sinb[(size_t)t * 64 + tid];
    qb[obase + tid]      = f2bf((x1 * c - x2 * s) * attscale);
    qb[obase + tid + 64] = f2bf((x1 * s + x2 * c) * attscale);
  } else {
    int i = tid - 64;
    float x1 = kp[i] * rk, x2 = kp[i + 64] * rk;
    float c = cosb[(size_t)t * 64 + i], s = sinb[(size_t)t * 64 + i];
    float o1 = x1 * c - x2 * s, o2 = x1 * s + x2 * c;
    float* kout = outbase + (size_t)T_SEQ * D_MODEL;
    kout[obase + i] = o1; kout[obase + i + 64] = o2;
    kb[obase + i] = f2bf(o1); kb[obase + i + 64] = f2bf(o2);
  }
  float vvv = vp[tid];
  float* vout = outbase + 2 * (size_t)T_SEQ * D_MODEL;
  vout[obase + tid] = vvv;
  vb[obase + tid] = f2bf(vvv);
}

// flash attention: one WG per (head, 128-row q block); 8 waves, 16 q rows each
__global__ __launch_bounds__(256) void attn_kernel(
    const unsigned short* __restrict__ qb, const unsigned short* __restrict__ kb,
    const unsigned short* __restrict__ vb, unsigned short* __restrict__ ob) {
  __shared__ unsigned short Ks[64 * 128];
  __shared__ unsigned short Vs[64 * 128];
  __shared__ unsigned short Ps[8][16 * 64];
  int head = blockIdx.x, qblk = blockIdx.y;
  int tid = threadIdx.x, wid = tid >> 5, lane = tid & 31;
  int qbase = qblk * 128, row0 = qbase + wid * 16;
  int half8 = ((lane >> 4) & 1) * 8, col_l = lane & 15;

  v16bf qfrag[4];
#pragma unroll
  for (int ks = 0; ks < 4; ++ks)
    qfrag[ks] = load_A16(qb + (size_t)row0 * D_MODEL + head * HDIM, 0, ks * 32, D_MODEL, lane);

  v8f zero = {0.f, 0.f, 0.f, 0.f, 0.f, 0.f, 0.f, 0.f};
  v8f o[8];
  float m_run[8], l_run[8];
#pragma unroll
  for (int i = 0; i < 8; ++i) { o[i] = zero; m_run[i] = -3.0e38f; l_run[i] = 0.f; }

  int row_last = row0 + 15;
  int jblocks = (qblk + 1) * 2;
  for (int jb = 0; jb < jblocks; ++jb) {
    int j0 = jb * 64;
#if HAVE_TDM
    if (wid == 0) {
      tdm_load_2d_bf16(lds_off(Ks), kb + (size_t)j0 * D_MODEL + head * HDIM, 128, 64,
                       (unsigned)D_MODEL, (unsigned)T_SEQ, (unsigned long long)D_MODEL);
      tdm_load_2d_bf16(lds_off(Vs), vb + (size_t)j0 * D_MODEL + head * HDIM, 128, 64,
                       (unsigned)D_MODEL, (unsigned)T_SEQ, (unsigned long long)D_MODEL);
      __builtin_amdgcn_s_wait_tensorcnt(0);
    }
#else
    copy_tile_coop(Ks, kb + (size_t)j0 * D_MODEL + head * HDIM, 64, 128, D_MODEL, tid, 256);
    copy_tile_coop(Vs, vb + (size_t)j0 * D_MODEL + head * HDIM, 64, 128, D_MODEL, tid, 256);
#endif
    __syncthreads();
    if (j0 <= row_last) {                      // wave-uniform causal skip
      v8f s[4];
#pragma unroll
      for (int nt = 0; nt < 4; ++nt) {
        v8f a = zero;
#pragma unroll
        for (int ks = 0; ks < 4; ++ks) {
          v16bf bk = load_rowB16(Ks, nt * 16 + col_l, ks * 32, 128, lane);
          a = wmma_bf16(qfrag[ks], bk, a);
        }
        s[nt] = a;
      }
#pragma unroll
      for (int nt = 0; nt < 4; ++nt)
#pragma unroll
        for (int r = 0; r < 8; ++r) {
          int qr = row0 + r + half8;
          int kc = j0 + nt * 16 + col_l;
          if (kc > qr) s[nt][r] = -3.0e38f;
        }
#pragma unroll
      for (int r = 0; r < 8; ++r) {
        float mx = fmaxf(fmaxf(s[0][r], s[1][r]), fmaxf(s[2][r], s[3][r]));
#pragma unroll
        for (int m = 8; m >= 1; m >>= 1) mx = fmaxf(mx, __shfl_xor(mx, m));
        float mtot = fmaxf(m_run[r], mx);
        float alpha = __expf(m_run[r] - mtot);
        m_run[r] = mtot;
        float rs = 0.f;
        int prow = (r + half8) * 64;
#pragma unroll
        for (int nt = 0; nt < 4; ++nt) {
          float p = __expf(s[nt][r] - mtot);
          rs += p;
          Ps[wid][prow + nt * 16 + col_l] = f2bf(p);
        }
#pragma unroll
        for (int m = 8; m >= 1; m >>= 1) rs += __shfl_xor(rs, m);
        l_run[r] = l_run[r] * alpha + rs;
#pragma unroll
        for (int nt8 = 0; nt8 < 8; ++nt8) o[nt8][r] = o[nt8][r] * alpha;
      }
#pragma unroll
      for (int ks2 = 0; ks2 < 2; ++ks2) {
        v16bf pa = load_A16(&Ps[wid][0], 0, ks2 * 32, 64, lane);
#pragma unroll
        for (int nt8 = 0; nt8 < 8; ++nt8) {
          v16bf bv = load_colB16(Vs, nt8 * 16 + col_l, ks2 * 32, 128, lane);
          o[nt8] = wmma_bf16(pa, bv, o[nt8]);
        }
      }
    }
    __syncthreads();
  }
#pragma unroll
  for (int r = 0; r < 8; ++r) {
    float inv = 1.0f / l_run[r];
    size_t base = (size_t)(row0 + r + half8) * D_MODEL + head * HDIM;
#pragma unroll
    for (int nt8 = 0; nt8 < 8; ++nt8)
      ob[base + nt8 * 16 + col_l] = f2bf(o[nt8][r] * inv);
  }
}

// ---------------------------------------------------------------- launcher
extern "C" void kernel_launch(void* const* d_in, const int* in_sizes, int n_in,
                              void* d_out, int out_size, void* d_ws, size_t ws_size,
                              hipStream_t stream) {
  (void)in_sizes; (void)n_in; (void)out_size; (void)ws_size;
  const float* x    = (const float*)d_in[0];
  const float* wqkv = (const float*)d_in[1];
  const float* wo   = (const float*)d_in[2];
  const float* cosb = (const float*)d_in[3];
  const float* sinb = (const float*)d_in[4];
  float* out = (float*)d_out;   // [out | k | v], each T*D f32

  char* ws = (char*)d_ws;
  unsigned short* xn   = (unsigned short*)(ws);                          //  8 MB
  unsigned short* wq_b = (unsigned short*)(ws + (8ull  << 20));          // 24 MB
  unsigned short* wo_b = (unsigned short*)(ws + (32ull << 20));          //  8 MB
  float*          qkv  = (float*)         (ws + (40ull << 20));          // 48 MB
  unsigned short* q_b  = (unsigned short*)(ws + (88ull << 20));          //  8 MB
  unsigned short* k_b  = (unsigned short*)(ws + (96ull << 20));          //  8 MB
  unsigned short* v_b  = (unsigned short*)(ws + (104ull << 20));         //  8 MB
  unsigned short* at_b = (unsigned short*)(ws + (112ull << 20));         //  8 MB

  // 1) weights -> bf16
  {
    int n2 = (3 * D_MODEL * D_MODEL) / 2;
    cast_bf16_kernel<<<(n2 + 255) / 256, 256, 0, stream>>>(wqkv, (unsigned*)wq_b, n2);
    int n2o = (D_MODEL * D_MODEL) / 2;
    cast_bf16_kernel<<<(n2o + 255) / 256, 256, 0, stream>>>(wo, (unsigned*)wo_b, n2o);
  }
  // 2) rmsnorm rows -> bf16
  rmsnorm_rows_kernel<<<T_SEQ, 256, 0, stream>>>(x, xn);
  // 3) qkv = xn @ Wqkv^T
  gemm_bf16_kernel<<<dim3(T_SEQ / 128, D3 / 128), 256, 0, stream>>>(
      xn, wq_b, qkv, nullptr, T_SEQ, D3, D_MODEL);
  // 4) per-head rmsnorm + rope; emits k,v outputs + bf16 q/k/v
  rope_kernel<<<dim3(NHEAD, T_SEQ), 128, 0, stream>>>(qkv, cosb, sinb, q_b, k_b, v_b, out);
  // 5) causal flash attention
  attn_kernel<<<dim3(NHEAD, T_SEQ / 128), 256, 0, stream>>>(q_b, k_b, v_b, at_b);
  // 6) out = x + attn @ Wo^T
  gemm_bf16_kernel<<<dim3(T_SEQ / 128, D_MODEL / 128), 256, 0, stream>>>(
      at_b, wo_b, out, x, T_SEQ, D_MODEL, D_MODEL);
}
